// CrossAttention_51539607552970
// MI455X (gfx1250) — matrix operand order repaired and checked by reading, not compiled, not run on hardware
//
#include <hip/hip_runtime.h>
#include <hip/hip_bf16.h>

// MI455X cross-attention: bf16 WMMA everywhere, fp32 accumulate, flash softmax,
// TDM (tensor_load_to_lds) for Q/K tile staging, global_prefetch in GEMM.

typedef __attribute__((ext_vector_type(16))) __bf16 v16bf;
typedef __attribute__((ext_vector_type(8)))  float  v8f;
typedef __attribute__((ext_vector_type(4)))  __bf16 v4bf;
typedef __attribute__((ext_vector_type(4)))  unsigned int u32x4;
typedef __attribute__((ext_vector_type(8)))  int i32x8;
typedef __attribute__((ext_vector_type(4)))  int i32x4;

#define B_    2
#define S_    2048
#define L_    2048
#define H_    16
#define KVH_  4
#define D_    128
#define HID_  2048
#define EPSF  1e-6f
#define SCALEF 0.08838834764831845f   // 1/sqrt(128)

// ---------------------------------------------------------------------------
// WMMA fragment gathers (ISA 7.12.2 16-bit layouts, wave32)
// ---------------------------------------------------------------------------
static __device__ __forceinline__ v16bf load_a_frag(const __bf16* base, int ld,
                                                    int m0, int k0, int lane) {
  const __bf16* p = base + (size_t)(m0 + (lane & 15)) * ld + (k0 + ((lane >> 4) << 3));
  v16bf a;
#pragma unroll
  for (int i = 0; i < 8; ++i) { a[i] = p[i]; a[8 + i] = p[16 + i]; }
  return a;
}

static __device__ __forceinline__ v16bf load_b_frag(const __bf16* base, int ld,
                                                    int n0, int k0, int lane) {
  const __bf16* p = base + (size_t)(n0 + (lane & 15)) * ld + (k0 + ((lane >> 4) << 3));
  v16bf b;
#pragma unroll
  for (int i = 0; i < 8; ++i) { b[i] = p[i]; b[8 + i] = p[16 + i]; }
  return b;
}

static __device__ __forceinline__ v8f wmma_bf16(v16bf a, v16bf b, v8f c) {
  return __builtin_amdgcn_wmma_f32_16x16x32_bf16(false, a, false, b, (short)0, c,
                                                 false, false);
}

// ---------------------------------------------------------------------------
// TDM: 2D tile load (bf16 elements) from global -> LDS via Tensor Data Mover.
// Descriptor per ISA 8.3/8.4: group0 = {count/lds_addr/global_addr/type},
// group1 = {data_size, tensor dims, tile dims, dim0 stride}. 2D tensor:
// groups 2/3 (and the trailing group) are zero. 6-arg builtin form
// (amdgpu-toolchain / clang-23).
// ---------------------------------------------------------------------------
static __device__ __forceinline__ void tdm_load_2d(unsigned lds_off,
                                                   const void* gptr,
                                                   unsigned tile_w, unsigned tile_h,
                                                   unsigned long long row_stride) {
  const unsigned long long ga = (unsigned long long)(uintptr_t)gptr;
  u32x4 g0;
  g0[0] = 1u;                                                // count=1 (user D#)
  g0[1] = lds_off;                                           // lds_addr (bytes)
  g0[2] = (unsigned)(ga & 0xFFFFFFFFu);                      // global_addr[31:0]
  g0[3] = (unsigned)((ga >> 32) & 0x01FFFFFFu) | (2u << 30); // ga[56:32] | type=2
  i32x8 g1;
  const unsigned tw = tile_w, th = tile_h;                   // tensor dims == tile
  g1[0] = (int)(1u << 16);                                   // data_size=1 -> 2B
  g1[1] = (int)((tw & 0xFFFFu) << 16);                       // tensor_dim0[15:0]
  g1[2] = (int)((tw >> 16) | ((th & 0xFFFFu) << 16));        // d0 hi | tensor_dim1 lo
  g1[3] = (int)((th >> 16) | (tile_w << 16));                // d1 hi | tile_dim0
  g1[4] = (int)(tile_h & 0xFFFFu);                           // tile_dim1 (dim2=0)
  g1[5] = (int)(row_stride & 0xFFFFFFFFull);                 // tensor_dim0_stride lo
  g1[6] = (int)((row_stride >> 32) & 0xFFFFull);             // stride hi (d1 stride=0)
  g1[7] = 0;
  const i32x4 z4 = {0, 0, 0, 0};
  const i32x8 z8 = {0, 0, 0, 0, 0, 0, 0, 0};
  __builtin_amdgcn_tensor_load_to_lds(g0, g1, z4, z4, z8, 0);
}

// ---------------------------------------------------------------------------
// fp32 -> bf16 elementwise
// ---------------------------------------------------------------------------
__global__ __launch_bounds__(256) void f2bf_kernel(const float* __restrict__ in,
                                                   __bf16* __restrict__ out, size_t n) {
  size_t i = ((size_t)blockIdx.x * 256 + threadIdx.x) * 4;
  if (i >= n) return;
  float4 v = *(const float4*)(in + i);
  v4bf o;
  o[0] = (__bf16)v.x; o[1] = (__bf16)v.y; o[2] = (__bf16)v.z; o[3] = (__bf16)v.w;
  *(v4bf*)(out + i) = o;
}

// ---------------------------------------------------------------------------
// Tiled bf16 GEMM: C[M,N] fp32 = A[M,K]bf16 @ B[K,N]bf16
// workgroup tile 128x128, K-tile 32, 8 waves in 2x4, wave tile 64x32.
// ---------------------------------------------------------------------------
__global__ __launch_bounds__(256) void gemm_bf16_kernel(const __bf16* __restrict__ A,
                                                        const __bf16* __restrict__ B,
                                                        float* __restrict__ C,
                                                        int M, int N, int K) {
  __shared__ __bf16 Alds[128][32];
  __shared__ __bf16 Blds[128][32];   // transposed: [N][K]

  const int bm = blockIdx.y * 128;
  const int bn = blockIdx.x * 128;
  const int tid  = threadIdx.x;
  const int wv   = tid >> 5;
  const int lane = tid & 31;
  const int wr = wv >> 2;   // 0..1  (64 rows)
  const int wc = wv & 3;    // 0..3  (32 cols)

  v8f acc[4][2];
#pragma unroll
  for (int i = 0; i < 4; ++i)
#pragma unroll
    for (int j = 0; j < 2; ++j)
#pragma unroll
      for (int e = 0; e < 8; ++e) acc[i][j][e] = 0.0f;

  const int arow = tid >> 1, acol = (tid & 1) * 16;
  const int bkk  = tid >> 3, bnn  = (tid & 7) * 16;

  for (int k0 = 0; k0 < K; k0 += 32) {
    // A tile: 128x32, 2 threads per row, 16 contiguous bf16 each (2x b128)
    {
      const __bf16* src = A + (size_t)(bm + arow) * K + k0 + acol;
      *(uint4*)&Alds[arow][acol]     = *(const uint4*)src;
      *(uint4*)&Alds[arow][acol + 8] = *(const uint4*)(src + 8);
    }
    // B tile: rows k0..k0+31, cols bn..bn+127, stored transposed
    {
      const __bf16* src = B + (size_t)(k0 + bkk) * N + bn + bnn;
#pragma unroll
      for (int i = 0; i < 16; ++i) Blds[bnn + i][bkk] = src[i];
    }
    // prefetch next K-tile into cache hierarchy (global_prefetch_b8)
    if (k0 + 32 < K) {
      __builtin_prefetch(A + (size_t)(bm + arow) * K + k0 + 32 + acol, 0, 0);
      __builtin_prefetch(B + (size_t)(k0 + 32 + bkk) * N + bn + bnn, 0, 0);
    }
    __syncthreads();

    v16bf bfr[2];
#pragma unroll
    for (int ni = 0; ni < 2; ++ni)
      bfr[ni] = load_b_frag(&Blds[0][0], 32, wc * 32 + ni * 16, 0, lane);
#pragma unroll
    for (int mi = 0; mi < 4; ++mi) {
      v16bf afr = load_a_frag(&Alds[0][0], 32, wr * 64 + mi * 16, 0, lane);
#pragma unroll
      for (int ni = 0; ni < 2; ++ni)
        acc[mi][ni] = wmma_bf16(afr, bfr[ni], acc[mi][ni]);
    }
    __syncthreads();
  }

  const int nl = lane & 15;
  const int mb = (lane >> 4) * 8;
#pragma unroll
  for (int mi = 0; mi < 4; ++mi)
#pragma unroll
    for (int ni = 0; ni < 2; ++ni) {
      float* dst = C + (size_t)(bm + wr * 64 + mi * 16 + mb) * N
                     + (bn + wc * 32 + ni * 16 + nl);
#pragma unroll
      for (int j = 0; j < 8; ++j) dst[(size_t)j * N] = acc[mi][ni][j];
    }
}

// ---------------------------------------------------------------------------
// Q RMSNorm: rows = B*S*H, one wave per row of D=128, fp32 in, bf16 out
// ---------------------------------------------------------------------------
__global__ __launch_bounds__(256) void qnorm_kernel(const float* __restrict__ qf,
                                                    const float* __restrict__ w,
                                                    __bf16* __restrict__ qb) {
  const int row  = blockIdx.x * 8 + (threadIdx.x >> 5);
  const int lane = threadIdx.x & 31;
  const float* src = qf + (size_t)row * D_;
  float v[4]; float ss = 0.0f;
#pragma unroll
  for (int i = 0; i < 4; ++i) { v[i] = src[lane + i * 32]; ss += v[i] * v[i]; }
#pragma unroll
  for (int off = 16; off > 0; off >>= 1) ss += __shfl_xor(ss, off, 32);
  const float r = rsqrtf(ss * (1.0f / D_) + EPSF);
#pragma unroll
  for (int i = 0; i < 4; ++i) {
    const int d = lane + i * 32;
    qb[(size_t)row * D_ + d] = (__bf16)(v[i] * r * w[d]);
  }
}

// ---------------------------------------------------------------------------
// KV split + K RMSNorm: ckvf rows [B*L, KVH*D*2] packed (..., kvh, d, 2)
// ---------------------------------------------------------------------------
__global__ __launch_bounds__(256) void kvnorm_kernel(const float* __restrict__ ckv,
                                                     const float* __restrict__ wk,
                                                     __bf16* __restrict__ kb,
                                                     __bf16* __restrict__ vb) {
  const int row  = blockIdx.x * 8 + (threadIdx.x >> 5);   // over B*L*KVH
  const int lane = threadIdx.x & 31;
  const int token = row >> 2;
  const int kvh   = row & 3;
  const float* base = ckv + (size_t)token * (KVH_ * D_ * 2) + kvh * (D_ * 2);
  float kv[4], vv[4]; float ss = 0.0f;
#pragma unroll
  for (int i = 0; i < 4; ++i) {
    const int d = lane + i * 32;
    float2 p = *(const float2*)(base + 2 * d);
    kv[i] = p.x; vv[i] = p.y; ss += p.x * p.x;
  }
#pragma unroll
  for (int off = 16; off > 0; off >>= 1) ss += __shfl_xor(ss, off, 32);
  const float r = rsqrtf(ss * (1.0f / D_) + EPSF);
#pragma unroll
  for (int i = 0; i < 4; ++i) {
    const int d = lane + i * 32;
    kb[(size_t)row * D_ + d] = (__bf16)(kv[i] * r * wk[d]);
    vb[(size_t)row * D_ + d] = (__bf16)vv[i];
  }
}

// ---------------------------------------------------------------------------
// Flash attention: grid (S/64, H, B), block 256. Q tile 64x128, KV tile 32.
// Q/K tiles staged by the Tensor Data Mover (wave 0 issues, TENSORcnt wait,
// then workgroup barrier publishes LDS to all waves). V transposed manually.
// ---------------------------------------------------------------------------
__global__ __launch_bounds__(256) void attn_kernel(const __bf16* __restrict__ Qb,
                                                   const __bf16* __restrict__ Kb,
                                                   const __bf16* __restrict__ Vb,
                                                   __bf16* __restrict__ Ob) {
  __shared__ __bf16 Qlds[64][128];
  __shared__ __bf16 Klds[32][128];  // [N=l][K=d]
  __shared__ __bf16 Vlds[128][32];  // [N=d][K=l] (transposed on load)
  __shared__ float  Slds[64][32];
  __shared__ __bf16 Plds[64][32];
  __shared__ float  rowm[64], rowl[64], rowalpha[64];

  const int qt = blockIdx.x, h = blockIdx.y, b = blockIdx.z;
  const int kvh = h / (H_ / KVH_);
  const int tid = threadIdx.x, wv = tid >> 5, lane = tid & 31;
  const int smt = wv >> 1, snt = wv & 1;   // score tile owner (4x2)
  const int wr  = wv >> 2, wc  = wv & 3;   // O tile owner (2x4)
  const int nl  = lane & 15, mb = (lane >> 4) * 8;

  // TDM: Q tile 64x128 (row stride H*D elements)
  if (wv == 0) {
    tdm_load_2d((unsigned)(uintptr_t)&Qlds[0][0],
                Qb + (((size_t)b * S_ + qt * 64) * H_ + h) * D_,
                /*tile_w=*/D_, /*tile_h=*/64,
                /*row_stride(elems)=*/(unsigned long long)H_ * D_);
  }
  if (tid < 64) { rowm[tid] = -1e30f; rowl[tid] = 0.0f; }
  if (wv == 0) __builtin_amdgcn_s_wait_tensorcnt(0);
  __syncthreads();

  // hoist this wave's Q fragments (row tile smt, four K-steps over D)
  v16bf qfrag[4];
#pragma unroll
  for (int kk = 0; kk < 4; ++kk)
    qfrag[kk] = load_a_frag(&Qlds[0][0], 128, smt * 16, kk * 32, lane);

  v8f o[2][2];
#pragma unroll
  for (int i = 0; i < 2; ++i)
#pragma unroll
    for (int j = 0; j < 2; ++j)
#pragma unroll
      for (int e = 0; e < 8; ++e) o[i][j][e] = 0.0f;

  for (int l0 = 0; l0 < L_; l0 += 32) {
    __syncthreads();   // protect Klds/Vlds/Plds from previous iteration readers
    // TDM: K tile 32x128 (row stride KVH*D elements)
    if (wv == 0) {
      tdm_load_2d((unsigned)(uintptr_t)&Klds[0][0],
                  Kb + (((size_t)b * L_ + l0) * KVH_ + kvh) * D_,
                  /*tile_w=*/D_, /*tile_h=*/32,
                  /*row_stride(elems)=*/(unsigned long long)KVH_ * D_);
    }
    // V tile transposed into [d][l] (manual: TDM has no transpose)
    {
      const int row = tid >> 3;          // l within tile: 0..31
      const int col = (tid & 7) * 16;    // d base
      const size_t gidx = (((size_t)b * L_ + l0 + row) * KVH_ + kvh) * D_ + col;
      __bf16 vtmp[16];
      *(uint4*)&vtmp[0] = *(const uint4*)(Vb + gidx);
      *(uint4*)&vtmp[8] = *(const uint4*)(Vb + gidx + 8);
#pragma unroll
      for (int i = 0; i < 16; ++i) Vlds[col + i][row] = vtmp[i];
    }
    if (wv == 0) __builtin_amdgcn_s_wait_tensorcnt(0);
    __syncthreads();

    // scores: each wave one 16x16 tile, K-dim = D in 4 steps
    {
      v8f sa;
#pragma unroll
      for (int e = 0; e < 8; ++e) sa[e] = 0.0f;
#pragma unroll
      for (int kk = 0; kk < 4; ++kk) {
        v16bf kf = load_b_frag(&Klds[0][0], 128, snt * 16, kk * 32, lane);
        sa = wmma_bf16(qfrag[kk], kf, sa);
      }
#pragma unroll
      for (int j = 0; j < 8; ++j)
        Slds[smt * 16 + mb + j][snt * 16 + nl] = sa[j] * SCALEF;
    }
    __syncthreads();

    // online softmax: one thread per query row
    if (tid < 64) {
      const float mold = rowm[tid];
      float mx = mold;
#pragma unroll
      for (int c = 0; c < 32; ++c) mx = fmaxf(mx, Slds[tid][c]);
      float sum = 0.0f;
#pragma unroll
      for (int c = 0; c < 32; ++c) {
        const float p = __expf(Slds[tid][c] - mx);
        Plds[tid][c] = (__bf16)p;
        sum += p;
      }
      const float alpha = __expf(mold - mx);
      rowalpha[tid] = alpha;
      rowm[tid] = mx;
      rowl[tid] = rowl[tid] * alpha + sum;
    }
    __syncthreads();

    // rescale O and accumulate P@V (K = 32, one WMMA per tile)
#pragma unroll
    for (int i = 0; i < 2; ++i) {
      float af[8];
#pragma unroll
      for (int j = 0; j < 8; ++j) af[j] = rowalpha[(2 * wr + i) * 16 + mb + j];
      v16bf pf = load_a_frag(&Plds[0][0], 32, (2 * wr + i) * 16, 0, lane);
#pragma unroll
      for (int ni = 0; ni < 2; ++ni) {
        v16bf vf = load_b_frag(&Vlds[0][0], 32, (2 * wc + ni) * 16, 0, lane);
#pragma unroll
        for (int j = 0; j < 8; ++j) o[i][ni][j] *= af[j];
        o[i][ni] = wmma_bf16(pf, vf, o[i][ni]);
      }
    }
  }

  // normalize and store: attn[b, q, h*D + d] bf16
#pragma unroll
  for (int i = 0; i < 2; ++i) {
    float invl[8];
#pragma unroll
    for (int j = 0; j < 8; ++j) invl[j] = 1.0f / rowl[(2 * wr + i) * 16 + mb + j];
#pragma unroll
    for (int ni = 0; ni < 2; ++ni) {
      const int d = (2 * wc + ni) * 16 + nl;
#pragma unroll
      for (int j = 0; j < 8; ++j) {
        const int q = qt * 64 + (2 * wr + i) * 16 + mb + j;
        Ob[((size_t)b * S_ + q) * HID_ + h * D_ + d] = (__bf16)(o[i][ni][j] * invl[j]);
      }
    }
  }
}

// ---------------------------------------------------------------------------
extern "C" void kernel_launch(void* const* d_in, const int* in_sizes, int n_in,
                              void* d_out, int out_size, void* d_ws, size_t ws_size,
                              hipStream_t stream) {
  const float* x   = (const float*)d_in[0];
  const float* c   = (const float*)d_in[1];
  const float* wq  = (const float*)d_in[2];
  const float* wkv = (const float*)d_in[3];
  const float* wo  = (const float*)d_in[4];
  const float* nqw = (const float*)d_in[5];
  const float* nkw = (const float*)d_in[6];
  float* out = (float*)d_out;

  const size_t NX  = (size_t)B_ * S_ * HID_;       // 8388608
  const size_t NC  = (size_t)B_ * L_ * HID_;       // 8388608
  const size_t NWQ = (size_t)HID_ * HID_;          // 4194304
  const size_t NWKV= (size_t)HID_ * KVH_ * D_ * 2; // 2097152
  const size_t NWO = (size_t)HID_ * HID_;          // 4194304

  char* ws = (char*)d_ws;
  size_t off = 0;
  __bf16* xb   = (__bf16*)(ws + off); off += NX * 2;
  __bf16* cb   = (__bf16*)(ws + off); off += NC * 2;
  __bf16* wqb  = (__bf16*)(ws + off); off += NWQ * 2;
  __bf16* wkvb = (__bf16*)(ws + off); off += NWKV * 2;
  __bf16* wob  = (__bf16*)(ws + off); off += NWO * 2;
  float*  qf   = (float*)(ws + off);  off += NX * 4;                        // [B*S, H*D]
  float*  ckvf = (float*)(ws + off);  off += (size_t)B_*L_*KVH_*D_*2 * 4;   // [B*L, KVH*D*2]
  __bf16* qb   = (__bf16*)(ws + off); off += NX * 2;
  __bf16* kb   = (__bf16*)(ws + off); off += (size_t)B_*L_*KVH_*D_ * 2;
  __bf16* vb   = (__bf16*)(ws + off); off += (size_t)B_*L_*KVH_*D_ * 2;
  __bf16* attnb= (__bf16*)(ws + off); off += NX * 2;

  // 1) fp32 -> bf16 conversions
  f2bf_kernel<<<NX  / 1024, 256, 0, stream>>>(x,   xb,   NX);
  f2bf_kernel<<<NC  / 1024, 256, 0, stream>>>(c,   cb,   NC);
  f2bf_kernel<<<NWQ / 1024, 256, 0, stream>>>(wq,  wqb,  NWQ);
  f2bf_kernel<<<NWKV/ 1024, 256, 0, stream>>>(wkv, wkvb, NWKV);
  f2bf_kernel<<<NWO / 1024, 256, 0, stream>>>(wo,  wob,  NWO);

  // 2) Q projection: [4096,2048] = [4096,2048] @ [2048,2048]
  gemm_bf16_kernel<<<dim3(HID_ / 128, (B_ * S_) / 128), 256, 0, stream>>>(
      xb, wqb, qf, B_ * S_, HID_, HID_);

  // 3) KV projection: [4096,1024] = [4096,2048] @ [2048,1024]
  gemm_bf16_kernel<<<dim3((KVH_ * D_ * 2) / 128, (B_ * L_) / 128), 256, 0, stream>>>(
      cb, wkvb, ckvf, B_ * L_, KVH_ * D_ * 2, HID_);

  // 4) RMSNorms + bf16 packing
  qnorm_kernel<<<(B_ * S_ * H_) / 8, 256, 0, stream>>>(qf, nqw, qb);
  kvnorm_kernel<<<(B_ * L_ * KVH_) / 8, 256, 0, stream>>>(ckvf, nkw, kb, vb);

  // 5) flash attention -> attnb [B*S, H*D] bf16
  attn_kernel<<<dim3(S_ / 64, H_, B_), 256, 0, stream>>>(qb, kb, vb, attnb);

  // 6) output projection: [4096,2048] fp32 = attnb @ wo
  gemm_bf16_kernel<<<dim3(HID_ / 128, (B_ * S_) / 128), 256, 0, stream>>>(
      attnb, wob, out, B_ * S_, HID_, HID_);
}
